// Attention_27650999451892
// MI455X (gfx1250) — compile-verified, matrix operand order checked by reading, not run on hardware
//
#include <hip/hip_runtime.h>
#include <hip/hip_bf16.h>
#include <math.h>

typedef _Float16 f16;
typedef __attribute__((ext_vector_type(16))) _Float16 v16h;
typedef __attribute__((ext_vector_type(8)))  _Float16 v8h;
typedef __attribute__((ext_vector_type(8)))  float    v8f;
typedef int v4i_ __attribute__((vector_size(16)));

#define S_    1024
#define HID_  2048
#define H_    16
#define KVH_  4
#define D_    128
#define C_    4096
#define KVD_  512
#define RSQRT_D 0.08838834764831845f

#define WMMA_F16(a,b,c) \
  __builtin_amdgcn_wmma_f32_16x16x32_f16(false,(a),false,(b),(short)0,(c),false,false)

// ---- CDNA5 async global->LDS copy (ASYNCcnt path), guarded fallback ----
#if defined(__AMDGCN__) && \
    __has_builtin(__builtin_amdgcn_global_load_async_to_lds_b128) && \
    __has_builtin(__builtin_amdgcn_s_wait_asynccnt)
#define ASYNC_LDS 1
#else
#define ASYNC_LDS 0
#endif

__device__ __forceinline__ void cp16(f16* l, const f16* g) {
#if ASYNC_LDS
  __builtin_amdgcn_global_load_async_to_lds_b128(
      (__attribute__((address_space(1))) v4i_*)(f16*)g,
      (__attribute__((address_space(3))) v4i_*)l, 0, 0);
#else
  *(v8h*)l = *(const v8h*)g;
#endif
}
__device__ __forceinline__ void cp_wait_all() {
#if ASYNC_LDS
  __builtin_amdgcn_s_wait_asynccnt(0);
#endif
}

// ---- WMMA fragment loaders (16-bit A/B layout per CDNA5 ISA 7.12.2) ----
__device__ inline v16h ld_frag_k(const f16* p, int half) {
  v16h f;
  const int k1 = half * 8;
#pragma unroll
  for (int e = 0; e < 8; ++e) f[e] = p[k1 + e];
#pragma unroll
  for (int e = 0; e < 8; ++e) f[8 + e] = p[16 + k1 + e];
  return f;
}
__device__ inline v16h ld_frag_s(const f16* p, int stride, int half) {
  v16h f;
  const int k1 = half * 8;
#pragma unroll
  for (int e = 0; e < 8; ++e) f[e] = p[(k1 + e) * stride];
#pragma unroll
  for (int e = 0; e < 8; ++e) f[8 + e] = p[(16 + k1 + e) * stride];
  return f;
}

// ---- block-level maxabs helper: shared reduce + 1 atomic per block ----
__device__ inline void block_maxabs(unsigned int local_bits,
                                    unsigned int* gout) {
  __shared__ unsigned int smax;
  if (threadIdx.x == 0) smax = 0u;
  __syncthreads();
  atomicMax(&smax, local_bits);
  __syncthreads();
  if (threadIdx.x == 0) atomicMax(gout, smax);
}

// ---------------- fq emulation: max-abs reduce + quantize ----------------
__global__ void init_scales_kernel(unsigned int* s) {
  if (threadIdx.x < 64) s[threadIdx.x] = 0u;
}

__global__ void maxabs_kernel(const float* __restrict__ x, int n,
                              unsigned int* __restrict__ out) {
  unsigned int m = 0u;
  for (int i = blockIdx.x * blockDim.x + threadIdx.x; i < n;
       i += gridDim.x * blockDim.x) {
    unsigned int b = __float_as_uint(fabsf(x[i]));
    m = (b > m) ? b : m;
  }
  block_maxabs(m, out);
}

__global__ void quantize_kernel(const float* __restrict__ x, int n,
                                const unsigned int* __restrict__ maxabs_bits,
                                float qmax, f16* __restrict__ out) {
  float ma = __uint_as_float(*maxabs_bits);
  float scale = fmaxf(ma, 1e-8f) / qmax;
  float inv = 1.0f / scale;
  for (int i = blockIdx.x * blockDim.x + threadIdx.x; i < n;
       i += gridDim.x * blockDim.x) {
    float q = rintf(x[i] * inv);
    q = fminf(fmaxf(q, -qmax - 1.0f), qmax);
    out[i] = (f16)(q * scale);
  }
}

// -------- tiled WMMA GEMM, double-buffered async LDS staging -----------
// C[M,N] = A[M,K] * B[N,K]^T + bias
__global__ __launch_bounds__(256) void gemm_f16_wmma(
    const f16* __restrict__ A, const f16* __restrict__ B,
    const float* __restrict__ bias, float* __restrict__ Cmat,
    int M, int N, int K) {
  __shared__ f16 As[2][128][40];
  __shared__ f16 Bs[2][128][40];
  const int bm = blockIdx.y * 128, bn = blockIdx.x * 128;
  const int tid = threadIdx.x;
  const int w = tid >> 5, lane = tid & 31;
  const int half = lane >> 4, r = lane & 15;
  const int wm = (w >> 1) * 32;  // 4 wave-rows x 32
  const int wn = (w & 1) * 64;   // 2 wave-cols x 64
  const v8f vz = {0, 0, 0, 0, 0, 0, 0, 0};
  v8f acc[2][4];
#pragma unroll
  for (int i = 0; i < 2; ++i)
#pragma unroll
    for (int j = 0; j < 4; ++j) acc[i][j] = vz;

  auto load_tile = [&](int k0, int buf) {
    // 128x32 halves -> 512 chunks of 8 halves (16B); 2 iters of 256 threads
    for (int i = tid; i < 128 * 4; i += 256) {
      int row = i >> 2, cc = (i & 3) << 3;
      cp16(&As[buf][row][cc], &A[(size_t)(bm + row) * K + k0 + cc]);
      cp16(&Bs[buf][row][cc], &B[(size_t)(bn + row) * K + k0 + cc]);
    }
  };

  load_tile(0, 0);
  int cur = 0;
  for (int k0 = 0; k0 < K; k0 += 32) {
    cp_wait_all();       // this wave's copies into buf[cur] complete
    __syncthreads();     // all waves' copies complete; prev reads done
    if (k0 + 32 < K) load_tile(k0 + 32, cur ^ 1);  // prefetch next tile
    v16h af[2], bf[4];
#pragma unroll
    for (int i = 0; i < 2; ++i)
      af[i] = ld_frag_k(&As[cur][wm + i * 16 + r][0], half);
#pragma unroll
    for (int j = 0; j < 4; ++j)
      bf[j] = ld_frag_k(&Bs[cur][wn + j * 16 + r][0], half);
#pragma unroll
    for (int i = 0; i < 2; ++i)
#pragma unroll
      for (int j = 0; j < 4; ++j) acc[i][j] = WMMA_F16(af[i], bf[j], acc[i][j]);
    cur ^= 1;
  }
#pragma unroll
  for (int i = 0; i < 2; ++i)
#pragma unroll
    for (int j = 0; j < 4; ++j)
#pragma unroll
      for (int v = 0; v < 8; ++v) {
        int row = bm + wm + i * 16 + half * 8 + v;
        int col = bn + wn + j * 16 + r;
        float b = bias ? bias[col] : 0.0f;
        Cmat[(size_t)row * N + col] = acc[i][j][v] + b;
      }
}

// ------- RoPE (+ fused maxabs of result when sc != nullptr) ------------
__global__ void rope_kernel(const float* __restrict__ in,
                            const float* __restrict__ cs,
                            const float* __restrict__ sn,
                            float* __restrict__ out, int nheads,
                            unsigned int* sc) {
  int total = S_ * nheads * (D_ / 2);
  unsigned int m = 0u;
  for (int idx = blockIdx.x * blockDim.x + threadIdx.x; idx < total;
       idx += gridDim.x * blockDim.x) {
    int dh = idx % (D_ / 2);
    int h = (idx / (D_ / 2)) % nheads;
    int s = idx / ((D_ / 2) * nheads);
    int base = s * nheads * D_ + h * D_;
    float x1 = in[base + dh], x2 = in[base + dh + D_ / 2];
    float c1 = cs[s * D_ + dh], c2 = cs[s * D_ + dh + D_ / 2];
    float s1 = sn[s * D_ + dh], s2 = sn[s * D_ + dh + D_ / 2];
    float o1 = x1 * c1 - x2 * s1;
    float o2 = x2 * c2 + x1 * s2;
    out[base + dh] = o1;
    out[base + dh + D_ / 2] = o2;
    unsigned int b1 = __float_as_uint(fabsf(o1));
    unsigned int b2 = __float_as_uint(fabsf(o2));
    m = (b1 > m) ? b1 : m;
    m = (b2 > m) ? b2 : m;
  }
  if (sc) block_maxabs(m, sc);
}

// -- cache concat + fused maxabs: dst[h][c][d] = c<C-S ? cache : fresh --
__global__ void concat_cache_kernel(const float* __restrict__ cache,
                                    const float* __restrict__ fresh,
                                    float* __restrict__ dst,
                                    unsigned int* sc) {
  int total = KVH_ * C_ * D_;
  unsigned int m = 0u;
  for (int idx = blockIdx.x * blockDim.x + threadIdx.x; idx < total;
       idx += gridDim.x * blockDim.x) {
    int d = idx % D_;
    int c = (idx / D_) % C_;
    int h = idx / (D_ * C_);
    float v;
    if (c < C_ - S_)
      v = cache[(size_t)h * C_ * D_ + (size_t)(c + S_) * D_ + d];
    else
      v = fresh[(size_t)(c - (C_ - S_)) * KVD_ + h * D_ + d];
    dst[idx] = v;
    unsigned int b = __float_as_uint(fabsf(v));
    m = (b > m) ? b : m;
  }
  block_maxabs(m, sc);
}

// ---------------- flash attention (online softmax, WMMA) ----------------
// grid: (S/128, H). block: 256 threads = 8 waves; wave w owns 16 query rows.
__global__ __launch_bounds__(256) void attn_kernel(
    const f16* __restrict__ qe, const f16* __restrict__ ck,
    const f16* __restrict__ cv, const float* __restrict__ mask,
    float* __restrict__ out, unsigned int* sc) {
  __shared__ f16 Ks[64][136];
  __shared__ f16 Vs[64][136];
  __shared__ f16 Pl[8][16][72];
  const int h = blockIdx.y;
  const int kvh = h >> 2;  // G = H/KVH = 4
  const int qbase = blockIdx.x * 128;
  const int tid = threadIdx.x;
  const int w = tid >> 5, lane = tid & 31;
  const int half = lane >> 4, r = lane & 15;
  const f16* ckh = ck + (size_t)kvh * C_ * D_;
  const f16* cvh = cv + (size_t)kvh * C_ * D_;

  v16h qf[4];
  {
    const f16* qp = qe + (size_t)(qbase + w * 16 + r) * HID_ + h * D_;
#pragma unroll
    for (int ks = 0; ks < 4; ++ks) qf[ks] = ld_frag_k(qp + ks * 32, half);
  }
  const v8f vz = {0, 0, 0, 0, 0, 0, 0, 0};
  v8f o_acc[8];
#pragma unroll
  for (int n = 0; n < 8; ++n) o_acc[n] = vz;
  float m_i[8], l_i[8];
#pragma unroll
  for (int v = 0; v < 8; ++v) { m_i[v] = -1e30f; l_i[v] = 0.0f; }

  for (int kt = 0; kt < C_; kt += 64) {
    __syncthreads();  // prev iteration's readers are done with Ks/Vs
    for (int i = tid; i < 64 * 16; i += 256) {  // 16B chunks
      int row = i >> 4, cc = (i & 15) << 3;
      cp16(&Ks[row][cc], &ckh[(size_t)(kt + row) * D_ + cc]);
      cp16(&Vs[row][cc], &cvh[(size_t)(kt + row) * D_ + cc]);
    }
    cp_wait_all();
    __syncthreads();
    // prefetch next tile's mask rows into L2 while we compute
    if (kt + 64 < C_) {
      int pq = qbase + w * 16 + half * 8;
      __builtin_prefetch(&mask[(size_t)pq * C_ + kt + 64 + r], 0, 1);
    }

    // S = Q(16x128) @ K_tile^T -> 16x64 per wave (16 WMMA)
    v8f sacc[4];
#pragma unroll
    for (int j = 0; j < 4; ++j) sacc[j] = vz;
#pragma unroll
    for (int ks = 0; ks < 4; ++ks)
#pragma unroll
      for (int j = 0; j < 4; ++j) {
        v16h b = ld_frag_k(&Ks[j * 16 + r][ks * 32], half);
        sacc[j] = WMMA_F16(qf[ks], b, sacc[j]);
      }

    // scale + additive mask
    float s[4][8];
#pragma unroll
    for (int j = 0; j < 4; ++j)
#pragma unroll
      for (int v = 0; v < 8; ++v) {
        int qrow = qbase + w * 16 + half * 8 + v;
        int kcol = kt + j * 16 + r;
        s[j][v] = sacc[j][v] * RSQRT_D + mask[(size_t)qrow * C_ + kcol];
      }

    // online softmax; each row's 64 cols live on the 16 lanes of this half
#pragma unroll
    for (int v = 0; v < 8; ++v) {
      float tm = s[0][v];
#pragma unroll
      for (int j = 1; j < 4; ++j) tm = fmaxf(tm, s[j][v]);
#pragma unroll
      for (int mm = 8; mm >= 1; mm >>= 1) tm = fmaxf(tm, __shfl_xor(tm, mm, 32));
      float mn = fmaxf(m_i[v], tm);
      float alpha = __expf(m_i[v] - mn);
      float ts = 0.0f;
#pragma unroll
      for (int j = 0; j < 4; ++j) {
        float p = __expf(s[j][v] - mn);
        s[j][v] = p;
        ts += p;
      }
#pragma unroll
      for (int mm = 8; mm >= 1; mm >>= 1) ts += __shfl_xor(ts, mm, 32);
      l_i[v] = l_i[v] * alpha + ts;
      m_i[v] = mn;
#pragma unroll
      for (int n = 0; n < 8; ++n) o_acc[n][v] *= alpha;
    }

    // stash P (16x64 f16) in this wave's LDS region, reload as A-fragments
#pragma unroll
    for (int j = 0; j < 4; ++j)
#pragma unroll
      for (int v = 0; v < 8; ++v)
        Pl[w][half * 8 + v][j * 16 + r] = (f16)s[j][v];

    // O += P(16x64) @ V_tile(64x128)  (16 WMMA)
#pragma unroll
    for (int ks2 = 0; ks2 < 2; ++ks2) {
      v16h pa = ld_frag_k(&Pl[w][r][ks2 * 32], half);
#pragma unroll
      for (int n = 0; n < 8; ++n) {
        v16h vb = ld_frag_s(&Vs[ks2 * 32][n * 16 + r], 136, half);
        o_acc[n] = WMMA_F16(pa, vb, o_acc[n]);
      }
    }
  }

  unsigned int mbits = 0u;
#pragma unroll
  for (int v = 0; v < 8; ++v) {
    float inv = 1.0f / fmaxf(l_i[v], 1e-30f);
    int qrow = qbase + w * 16 + half * 8 + v;
#pragma unroll
    for (int n = 0; n < 8; ++n) {
      float o = o_acc[n][v] * inv;
      out[(size_t)qrow * HID_ + h * D_ + n * 16 + r] = o;
      unsigned int b = __float_as_uint(fabsf(o));
      mbits = (b > mbits) ? b : mbits;
    }
  }
  block_maxabs(mbits, sc);  // fused fq(out,8) max-abs
}

// ---------------------------- launcher ----------------------------------
extern "C" void kernel_launch(void* const* d_in, const int* in_sizes, int n_in,
                              void* d_out, int out_size, void* d_ws,
                              size_t ws_size, hipStream_t stream) {
  (void)in_sizes; (void)n_in; (void)out_size; (void)ws_size;
  const float* x       = (const float*)d_in[0];
  const float* cs      = (const float*)d_in[1];
  const float* sn      = (const float*)d_in[2];
  const float* cache_k = (const float*)d_in[3];
  const float* cache_v = (const float*)d_in[4];
  const float* mask    = (const float*)d_in[5];
  const float* Wq      = (const float*)d_in[6];
  const float* bq      = (const float*)d_in[7];
  const float* Wk      = (const float*)d_in[8];
  const float* bk      = (const float*)d_in[9];
  const float* Wv      = (const float*)d_in[10];
  const float* bv      = (const float*)d_in[11];
  const float* Wo      = (const float*)d_in[12];
  float* out = (float*)d_out;

  const size_t NX = (size_t)S_ * HID_;          // 2M
  const size_t NWQ = (size_t)HID_ * HID_;       // 4M
  const size_t NWK = (size_t)KVD_ * HID_;       // 1M
  const size_t NKV = (size_t)S_ * KVD_;         // 512K
  const size_t NCK = (size_t)KVH_ * C_ * D_;    // 2M

  // d_out: [out 2M | k_out 512K | v_out 512K]
  float* k_out = out + NX;
  float* v_out = out + NX + NKV;

  char* ws = (char*)d_ws;
  size_t off = 0;
  auto alloc = [&](size_t bytes) {
    char* p = ws + off;
    off += (bytes + 255) & ~(size_t)255;
    return p;
  };
  unsigned int* scales = (unsigned int*)alloc(256);
  f16* x_q   = (f16*)alloc(NX * 2);
  f16* Wq_q  = (f16*)alloc(NWQ * 2);
  f16* Wk_q  = (f16*)alloc(NWK * 2);
  f16* Wv_q  = (f16*)alloc(NWK * 2);
  f16* Wo_q  = (f16*)alloc(NWQ * 2);
  float* q_f   = (float*)alloc(NX * 4);
  float* qe_f  = (float*)alloc(NX * 4);
  float* ke_f  = (float*)alloc(NKV * 4);
  float* ck_f  = (float*)alloc(NCK * 4);
  float* cv_f  = (float*)alloc(NCK * 4);
  f16* qe_q  = (f16*)alloc(NX * 2);
  f16* ck_q  = (f16*)alloc(NCK * 2);
  f16* cv_q  = (f16*)alloc(NCK * 2);
  float* ao_f  = (float*)alloc(NX * 4);
  f16* ao_q  = (f16*)alloc(NX * 2);

  const int T = 256;
  const dim3 RG(512);  // grid-stride reduce/quantize blocks

  init_scales_kernel<<<1, 64, 0, stream>>>(scales);

  // per-tensor max-abs for fq() scales of raw inputs
  maxabs_kernel<<<RG, T, 0, stream>>>(x,  (int)NX,  scales + 0);
  maxabs_kernel<<<RG, T, 0, stream>>>(Wq, (int)NWQ, scales + 1);
  maxabs_kernel<<<RG, T, 0, stream>>>(Wk, (int)NWK, scales + 2);
  maxabs_kernel<<<RG, T, 0, stream>>>(Wv, (int)NWK, scales + 3);
  maxabs_kernel<<<RG, T, 0, stream>>>(Wo, (int)NWQ, scales + 4);

  quantize_kernel<<<RG, T, 0, stream>>>(x,  (int)NX,  scales + 0, 127.0f, x_q);
  quantize_kernel<<<RG, T, 0, stream>>>(Wq, (int)NWQ, scales + 1, 127.0f, Wq_q);
  quantize_kernel<<<RG, T, 0, stream>>>(Wk, (int)NWK, scales + 2, 127.0f, Wk_q);
  quantize_kernel<<<RG, T, 0, stream>>>(Wv, (int)NWK, scales + 3, 127.0f, Wv_q);
  quantize_kernel<<<RG, T, 0, stream>>>(Wo, (int)NWQ, scales + 4, 127.0f, Wo_q);

  // projections (K/V land directly in d_out as k_out/v_out)
  gemm_f16_wmma<<<dim3(HID_ / 128, S_ / 128), T, 0, stream>>>(
      x_q, Wq_q, bq, q_f, S_, HID_, HID_);
  gemm_f16_wmma<<<dim3(KVD_ / 128, S_ / 128), T, 0, stream>>>(
      x_q, Wk_q, bk, k_out, S_, KVD_, HID_);
  gemm_f16_wmma<<<dim3(KVD_ / 128, S_ / 128), T, 0, stream>>>(
      x_q, Wv_q, bv, v_out, S_, KVD_, HID_);

  // RoPE (q's fq(.,8) max-abs fused); k's scale comes from ck concat
  rope_kernel<<<RG, T, 0, stream>>>(q_f, cs, sn, qe_f, H_, scales + 5);
  rope_kernel<<<RG, T, 0, stream>>>(k_out, cs, sn, ke_f, KVH_, nullptr);

  // cache concat with fused max-abs (ck 16-bit scale, cv 8-bit scale)
  concat_cache_kernel<<<RG, T, 0, stream>>>(cache_k, ke_f, ck_f, scales + 6);
  concat_cache_kernel<<<RG, T, 0, stream>>>(cache_v, v_out, cv_f, scales + 7);

  quantize_kernel<<<RG, T, 0, stream>>>(qe_f, (int)NX,  scales + 5, 127.0f, qe_q);
  quantize_kernel<<<RG, T, 0, stream>>>(ck_f, (int)NCK, scales + 6, 32767.0f, ck_q);
  quantize_kernel<<<RG, T, 0, stream>>>(cv_f, (int)NCK, scales + 7, 127.0f, cv_q);

  // flash attention (fused fq(out,8) max-abs into scales+8)
  attn_kernel<<<dim3(S_ / 128, H_), T, 0, stream>>>(qe_q, ck_q, cv_q, mask,
                                                    ao_f, scales + 8);

  // output projection
  quantize_kernel<<<RG, T, 0, stream>>>(ao_f, (int)NX, scales + 8, 127.0f, ao_q);
  gemm_f16_wmma<<<dim3(HID_ / 128, S_ / 128), T, 0, stream>>>(
      ao_q, Wo_q, nullptr, out, S_, HID_, HID_);
}
